// experts_module_list_81389630259223
// MI455X (gfx1250) — matrix operand order repaired and checked by reading, not compiled, not run on hardware
//
#include <hip/hip_runtime.h>
#include <hip/hip_bf16.h>

#define NTOK 4096
#define DIM  4096
#define NEXP 64
#define EMB  128
#define MOUT 11008
#define INV_T 2.5f       // 1/0.4
#define BASE_OFF 3.0f
#define LN_EPS 1e-5f

typedef __attribute__((ext_vector_type(16))) __bf16          bf16x16;
typedef __attribute__((ext_vector_type(16))) unsigned short  u16x16;
typedef __attribute__((ext_vector_type(8)))  unsigned short  u16x8;
typedef __attribute__((ext_vector_type(8)))  float           f32x8;

// Workspace layout (bytes)
#define OFF_XB      ((size_t)0)            // X bf16:  4096*4096*2 = 33554432
#define OFF_WRB     ((size_t)33554432)     // Wr bf16: 64*4096*2   = 524288
#define OFF_WDB     ((size_t)34078720)     // Wd bf16: 11008*128*2 = 2818048
#define OFF_LOGITS  ((size_t)36896768)     // 4 * 4096*64 * 4      = 4194304
#define OFF_HB      ((size_t)41091072)     // h bf16:  4096*128*2  = 1048576

// fp32 -> bf16 (round-to-nearest-even) as raw bits.
__device__ __forceinline__ unsigned short f2bf_bits(float f) {
  union { float f; unsigned u; } v; v.f = f;
  unsigned r = v.u + 0x7FFFu + ((v.u >> 16) & 1u);
  return (unsigned short)(r >> 16);
}

// Build a 16-element bf16 fragment from two 16-byte chunks (global or LDS).
__device__ __forceinline__ bf16x16 load_frag2(const unsigned short* p0,
                                              const unsigned short* p1) {
  union { u16x16 v; u16x8 h[2]; } f;
  f.h[0] = *(const u16x8*)p0;
  f.h[1] = *(const u16x8*)p1;
  return __builtin_bit_cast(bf16x16, f.v);
}

// Counter-based stateless RNG (deterministic across replays).
__device__ __forceinline__ unsigned hash_u32(unsigned x) {
  x ^= x >> 16; x *= 0x7FEB352Du;
  x ^= x >> 15; x *= 0x846CA68Bu;
  x ^= x >> 16; return x;
}
__device__ __forceinline__ float rng_u01(unsigned idx, unsigned stream) {
  unsigned h = hash_u32(idx * 0x9E3779B9u + stream * 0x85EBCA6Bu + 0x12345u);
  return ((float)(h >> 8) + 0.5f) * (1.0f / 16777216.0f);  // (0,1) strictly
}

// ---------------------------------------------------------------------------
// Kernel 0: bulk fp32 -> bf16 conversion, 8 elements/thread, fully vectorized.
// ---------------------------------------------------------------------------
__global__ void __launch_bounds__(256) f32_to_bf16_kernel(
    const float* __restrict__ src, unsigned short* __restrict__ dst, unsigned n8) {
  unsigned i = blockIdx.x * 256 + threadIdx.x;
  if (i >= n8) return;
  const float4 a = ((const float4*)src)[i * 2];
  const float4 b = ((const float4*)src)[i * 2 + 1];
  u16x8 o;
  o[0] = f2bf_bits(a.x); o[1] = f2bf_bits(a.y);
  o[2] = f2bf_bits(a.z); o[3] = f2bf_bits(a.w);
  o[4] = f2bf_bits(b.x); o[5] = f2bf_bits(b.y);
  o[6] = f2bf_bits(b.z); o[7] = f2bf_bits(b.w);
  ((u16x8*)dst)[i] = o;
}

// ---------------------------------------------------------------------------
// Kernel 1: router logits  X[4096,4096] @ Wr[64,4096]^T -> 4 K-sliced partials
// Block = 4 waves = one 16-token M-tile, each wave owns a 1024-wide K slice.
// ---------------------------------------------------------------------------
__global__ void __launch_bounds__(128) router_gemm_kernel(
    const unsigned short* __restrict__ xb, const unsigned short* __restrict__ wrb,
    float* __restrict__ logits_part) {
  const int lane   = threadIdx.x & 31;
  const int kslice = threadIdx.x >> 5;      // 0..3
  const int mtile  = blockIdx.x;            // 0..255
  const int half   = lane >> 4;
  const int l16    = lane & 15;
  const int row    = mtile * 16 + l16;
  const unsigned short* arow = xb + (size_t)row * DIM;

  f32x8 acc[4];
  #pragma unroll
  for (int n = 0; n < 4; ++n)
    #pragma unroll
    for (int j = 0; j < 8; ++j) acc[n][j] = 0.0f;

  const int kbeg = kslice * (DIM / 4);
  const int kend = kbeg + (DIM / 4);
  for (int k0 = kbeg; k0 < kend; k0 += 32) {
    bf16x16 a = load_frag2(arow + k0 + half * 8, arow + k0 + 16 + half * 8);
    #pragma unroll
    for (int n = 0; n < 4; ++n) {
      const unsigned short* brow =
          wrb + (size_t)(n * 16 + l16) * DIM + k0 + half * 8;
      bf16x16 b = load_frag2(brow, brow + 16);
      acc[n] = __builtin_amdgcn_wmma_f32_16x16x32_bf16(
          false, a, false, b, (short)0, acc[n], false, false);
    }
  }

  float* lp = logits_part + (size_t)kslice * NTOK * NEXP;
  #pragma unroll
  for (int n = 0; n < 4; ++n)
    #pragma unroll
    for (int j = 0; j < 8; ++j)
      lp[(size_t)(mtile * 16 + j + half * 8) * NEXP + n * 16 + l16] = acc[n][j];
}

// ---------------------------------------------------------------------------
// Kernel 2: sum K partials, gumbel-softmax routing (ST), gather + LN + GELU.
// One thread per token; z kept in a fully unrolled register array.
// ---------------------------------------------------------------------------
__global__ void __launch_bounds__(256) route_ln_gelu_kernel(
    const float* __restrict__ logits_part, const float* __restrict__ rnn_state,
    const float* __restrict__ ln_w,        const float* __restrict__ ln_b,
    float* __restrict__ out_router,        unsigned short* __restrict__ h_out) {
  const int t = blockIdx.x * blockDim.x + threadIdx.x;
  if (t >= NTOK) return;
  const float* p0 = logits_part + (size_t)t * NEXP;
  const float* p1 = p0 + (size_t)1 * NTOK * NEXP;
  const float* p2 = p0 + (size_t)2 * NTOK * NEXP;
  const float* p3 = p0 + (size_t)3 * NTOK * NEXP;

  float zbuf[NEXP];
  float zmax = -1e30f; int amax = 0;
  #pragma unroll
  for (int e = 0; e < NEXP; ++e) {
    float l = (p0[e] + p1[e]) + (p2[e] + p3[e]);
    float u = rng_u01((unsigned)(t * NEXP + e), 0u);
    float g = -__logf(-__logf(u + 1e-20f) + 1e-20f);
    float z = (l + g) * INV_T;
    zbuf[e] = z;
    if (z > zmax) { zmax = z; amax = e; }   // first max wins, as jnp.argmax
  }
  float sum = 0.0f;
  #pragma unroll
  for (int e = 0; e < NEXP; ++e) sum += __expf(zbuf[e] - zmax);
  const float rsum = 1.0f / sum;

  float r_gate = 1.0f;
  #pragma unroll
  for (int e = 0; e < NEXP; ++e) {
    float s = __expf(zbuf[e] - zmax) * rsum;
    float hard = (e == amax) ? 1.0f : 0.0f;
    float rv = (hard - s) + s;              // off-argmax entries are exactly 0
    out_router[(size_t)t * NEXP + e] = rv;
    if (e == amax) r_gate = rv;
  }

  // routed = r_gate * rnn_state[amax]; LN(128); exact-erf GELU; emit bf16
  const float* emb = rnn_state + (size_t)amax * EMB;
  float mu = 0.0f, m2 = 0.0f;
  for (int c = 0; c < EMB; ++c) { float v = r_gate * emb[c]; mu += v; m2 += v * v; }
  mu *= (1.0f / EMB);
  float var  = m2 * (1.0f / EMB) - mu * mu;
  float rstd = rsqrtf(var + LN_EPS);
  for (int c = 0; c < EMB; ++c) {
    float v  = r_gate * emb[c];
    float hn = (v - mu) * rstd * ln_w[c] + ln_b[c];
    float ge = 0.5f * hn * (1.0f + erff(hn * 0.70710678118f));
    h_out[(size_t)t * EMB + c] = f2bf_bits(ge);
  }
}

// ---------------------------------------------------------------------------
// Kernel 3: decode GEMM h[4096,128] @ Wd[11008,128]^T + gumbel-sigmoid ST.
// Block = 4 waves sharing one 64-column Wd panel staged in LDS (padded rows).
// Each wave: one 16x64 output tile, 16 WMMAs, B fragments via ds_load_b128.
// ---------------------------------------------------------------------------
#define BPAD 136   // 128 + 8 bf16 padding to spread LDS banks

__global__ void __launch_bounds__(128) decode_gemm_kernel(
    const unsigned short* __restrict__ hb, const unsigned short* __restrict__ wdb,
    float* __restrict__ out) {
  __shared__ unsigned short ldsB[64 * BPAD];

  const int lane   = threadIdx.x & 31;
  const int wid    = threadIdx.x >> 5;
  const int ngroup = blockIdx.x >> 6;       // 0..171 (64-column panel of Wd)
  const int mquad  = blockIdx.x & 63;       // 0..63
  const int mtile  = mquad * 4 + wid;       // 0..255

  // Cooperative stage: 64 Wd rows x 128 bf16 -> LDS (2 threads per row).
  {
    const int r  = threadIdx.x >> 1;        // 0..63
    const int h2 = threadIdx.x & 1;         // 0..1
    const unsigned short* src = wdb + (size_t)(ngroup * 64 + r) * EMB + h2 * 64;
    unsigned short* dst = ldsB + r * BPAD + h2 * 64;
    #pragma unroll
    for (int i = 0; i < 8; ++i)
      *(u16x8*)(dst + i * 8) = *(const u16x8*)(src + i * 8);
  }
  __syncthreads();

  const int half = lane >> 4;
  const int l16  = lane & 15;
  const int row  = mtile * 16 + l16;

  // A fragments for all of K=128 (reused across the 4 N-tiles)
  bf16x16 afrag[4];
  #pragma unroll
  for (int s = 0; s < 4; ++s) {
    const unsigned short* hr = hb + (size_t)row * EMB + s * 32 + half * 8;
    afrag[s] = load_frag2(hr, hr + 16);
  }

  f32x8 acc[4];
  #pragma unroll
  for (int n = 0; n < 4; ++n)
    #pragma unroll
    for (int j = 0; j < 8; ++j) acc[n][j] = 0.0f;

  #pragma unroll
  for (int n = 0; n < 4; ++n) {
    #pragma unroll
    for (int s = 0; s < 4; ++s) {
      const unsigned short* bp = ldsB + (n * 16 + l16) * BPAD + s * 32 + half * 8;
      bf16x16 b = load_frag2(bp, bp + 16);
      acc[n] = __builtin_amdgcn_wmma_f32_16x16x32_bf16(
          false, afrag[s], false, b, (short)0, acc[n], false, false);
    }
  }

  // Fused ST gumbel-sigmoid epilogue
  #pragma unroll
  for (int n = 0; n < 4; ++n) {
    const int col = ngroup * 64 + n * 16 + l16;
    #pragma unroll
    for (int j = 0; j < 8; ++j) {
      const int r = mtile * 16 + j + half * 8;
      const size_t idx = (size_t)r * MOUT + col;
      float pre  = acc[n][j];
      float u    = rng_u01((unsigned)idx, 1u);
      float ex   = -__logf(u);              // exponential sample
      float gum  = -__logf(ex);
      float zz   = (pre + gum + BASE_OFF) * INV_T;
      float s    = 1.0f / (1.0f + __expf(-zz));
      float hard = (s >= 0.5f) ? 1.0f : 0.0f;
      out[idx] = (hard - s) + s;
    }
  }
}

// ---------------------------------------------------------------------------
extern "C" void kernel_launch(void* const* d_in, const int* in_sizes, int n_in,
                              void* d_out, int out_size, void* d_ws, size_t ws_size,
                              hipStream_t stream) {
  const float* x         = (const float*)d_in[0];  // [2,2048,4096]
  const float* rnn_state = (const float*)d_in[1];  // [64,128]
  const float* Wr        = (const float*)d_in[2];  // [64,4096]
  const float* Wd        = (const float*)d_in[3];  // [11008,128]
  const float* ln_w      = (const float*)d_in[4];  // [128]
  const float* ln_b      = (const float*)d_in[5];  // [128]

  float* out_binary = (float*)d_out;                      // [4096,11008]
  float* out_router = out_binary + (size_t)NTOK * MOUT;   // [4096,64]

  char* ws = (char*)d_ws;
  unsigned short* xb     = (unsigned short*)(ws + OFF_XB);
  unsigned short* wrb    = (unsigned short*)(ws + OFF_WRB);
  unsigned short* wdb    = (unsigned short*)(ws + OFF_WDB);
  float*          logits = (float*)(ws + OFF_LOGITS);     // [4][4096][64]
  unsigned short* hb     = (unsigned short*)(ws + OFF_HB);

  // K0: one-time fp32 -> bf16 conversion of all GEMM operands
  f32_to_bf16_kernel<<<(NTOK * DIM / 8) / 256, 256, 0, stream>>>(x, xb, NTOK * DIM / 8);
  f32_to_bf16_kernel<<<(NEXP * DIM / 8) / 256, 256, 0, stream>>>(Wr, wrb, NEXP * DIM / 8);
  f32_to_bf16_kernel<<<(MOUT * EMB / 8) / 256, 256, 0, stream>>>(Wd, wdb, MOUT * EMB / 8);

  // K1: 256 blocks x 4 waves (M-tile x K-slice) = 1024 waves
  router_gemm_kernel<<<256, 128, 0, stream>>>(xb, wrb, logits);

  // K2: one thread per token
  route_ln_gelu_kernel<<<NTOK / 256, 256, 0, stream>>>(
      logits, rnn_state, ln_w, ln_b, out_router, hb);

  // K3: 172 Wd panels x 64 m-quads = 11008 blocks, 4 waves each
  decode_gemm_kernel<<<(MOUT / 64) * 64, 128, 0, stream>>>(hb, wdb, out_binary);
}
